// GAT_72043781423168
// MI455X (gfx1250) — matrix operand order repaired and checked by reading, not compiled, not run on hardware
//
#include <hip/hip_runtime.h>
#include <hip/hip_bf16.h>

typedef __attribute__((ext_vector_type(16))) _Float16 v16h;
typedef __attribute__((ext_vector_type(8)))  _Float16 v8h;
typedef __attribute__((ext_vector_type(4)))  _Float16 v4h;
typedef __attribute__((ext_vector_type(8)))  float    v8f;

#define GAT_N   50000
#define GAT_E   800000
#define GAT_ET  (GAT_E + GAT_N)
#define GAT_IN  256
#define GAT_HID 256
#define GAT_OUT 128

// ---------------------------------------------------------------------------
// f32 -> f16 bulk convert (vectorized: b128 load, b64 store)
// ---------------------------------------------------------------------------
__global__ void gat_cvt_f16(const float* __restrict__ in,
                            _Float16* __restrict__ out, int n4) {
  int i = blockIdx.x * blockDim.x + threadIdx.x;
  if (i >= n4) return;
  float4 v = ((const float4*)in)[i];
  v4h h;
  h[0] = (_Float16)v.x; h[1] = (_Float16)v.y;
  h[2] = (_Float16)v.z; h[3] = (_Float16)v.w;
  ((v4h*)out)[i] = h;
}

// f32 W[K][Nc] -> f16 Wt[Nc][K]  (tiny: <=64K elements)
__global__ void gat_cvt_transpose(const float* __restrict__ W,
                                  _Float16* __restrict__ Wt, int K, int Nc) {
  int idx = blockIdx.x * blockDim.x + threadIdx.x;
  if (idx >= K * Nc) return;
  int k = idx / Nc, c = idx % Nc;
  Wt[(size_t)c * K + k] = (_Float16)W[idx];
}

// ---------------------------------------------------------------------------
// GEMM: C[M,Nc] = A[M,K] * B[K,Nc], A f16 row-major, B given as Bt[Nc][K] f16.
// 256 threads = 8 waves; block tile 128x64; each wave owns a 16x64 strip.
// Fragments are contiguous 16B runs -> two global_load_b128 each, no LDS.
// Software-pipelined: K-step k+1 loads issue before K-step k's WMMAs.
// ---------------------------------------------------------------------------
__device__ __forceinline__ v16h gat_ldfrag(const _Float16* __restrict__ p) {
  v8h x0 = *(const v8h*)(p);
  v8h x1 = *(const v8h*)(p + 8);
  return __builtin_shufflevector(x0, x1, 0, 1, 2, 3, 4, 5, 6, 7,
                                         8, 9, 10, 11, 12, 13, 14, 15);
}

__global__ __launch_bounds__(256)
void gat_gemm_wmma(const _Float16* __restrict__ A,
                   const _Float16* __restrict__ Bt,
                   float* __restrict__ C, int M, int K, int Nc) {
  const int tid  = threadIdx.x;
  const int wave = tid >> 5;
  const int lane = tid & 31;
  const int lo   = lane & 15;
  const int hi   = lane >> 4;
  const int blockRow = blockIdx.x * 128;
  const int blockCol = blockIdx.y * 64;

  int rowA = blockRow + wave * 16 + lo;
  if (rowA > M - 1) rowA = M - 1;              // clamp: no divergent loads
  const _Float16* __restrict__ arow = A + (size_t)rowA * K;
  const _Float16* __restrict__ bcol[4];
  #pragma unroll
  for (int n = 0; n < 4; ++n)
    bcol[n] = Bt + (size_t)(blockCol + n * 16 + lo) * K + hi * 16;

  v8f acc[4];
  #pragma unroll
  for (int n = 0; n < 4; ++n)
    #pragma unroll
    for (int v = 0; v < 8; ++v) acc[n][v] = 0.0f;

  // prologue: fragments for k0 = 0
  // A fragment 16x32: lane row = lo; halves k = k0 + hi*16 + 0..15
  // B fragment 32x16: lane col = lo; halves k = k0 + hi*16 + 0..15
  v16h aCur = gat_ldfrag(arow + hi * 16);       // spans k0..k0+31 via two runs
  v16h bCur0 = gat_ldfrag(bcol[0]);
  v16h bCur1 = gat_ldfrag(bcol[1]);
  v16h bCur2 = gat_ldfrag(bcol[2]);
  v16h bCur3 = gat_ldfrag(bcol[3]);

  #pragma unroll
  for (int k0 = 0; k0 < GAT_HID; k0 += 32) {   // K == 256 for both layers
    v16h aNxt, bNxt0, bNxt1, bNxt2, bNxt3;
    if (k0 + 32 < GAT_HID) {
      aNxt  = gat_ldfrag(arow + (k0 + 32) + hi * 16);
      bNxt0 = gat_ldfrag(bcol[0] + (k0 + 32));
      bNxt1 = gat_ldfrag(bcol[1] + (k0 + 32));
      bNxt2 = gat_ldfrag(bcol[2] + (k0 + 32));
      bNxt3 = gat_ldfrag(bcol[3] + (k0 + 32));
    }
    acc[0] = __builtin_amdgcn_wmma_f32_16x16x32_f16(
        false, aCur, false, bCur0, (short)0, acc[0], false, false);
    acc[1] = __builtin_amdgcn_wmma_f32_16x16x32_f16(
        false, aCur, false, bCur1, (short)0, acc[1], false, false);
    acc[2] = __builtin_amdgcn_wmma_f32_16x16x32_f16(
        false, aCur, false, bCur2, (short)0, acc[2], false, false);
    acc[3] = __builtin_amdgcn_wmma_f32_16x16x32_f16(
        false, aCur, false, bCur3, (short)0, acc[3], false, false);
    aCur = aNxt;
    bCur0 = bNxt0; bCur1 = bNxt1; bCur2 = bNxt2; bCur3 = bNxt3;
  }

  // store: D layout — VGPR v: row = v + 8*hi, col = lane&15
  int rowBase = blockRow + wave * 16;
  #pragma unroll
  for (int n = 0; n < 4; ++n) {
    int col = blockCol + n * 16 + lo;
    #pragma unroll
    for (int v = 0; v < 8; ++v) {
      int row = rowBase + v + hi * 8;
      if (row < M) C[(size_t)row * Nc + col] = acc[n][v];
    }
  }
}

// A-fragment wrinkle: the two 16-byte runs of an A fragment live at
// arow + k0 + hi*16 and arow + k0 + hi*16 + 8; gat_ldfrag(arow + k0 + hi*16)
// loads p and p+16B, which is exactly those two runs. (For B the two runs are
// contiguous 32B at bcol + k0.)

// ---------------------------------------------------------------------------
// Per-node alpha dot products: one wave32 per node.
// ---------------------------------------------------------------------------
__global__ void gat_alpha(const float* __restrict__ g,
                          const float* __restrict__ asrc,
                          const float* __restrict__ adst,
                          float* __restrict__ outS, float* __restrict__ outD,
                          int N, int F) {
  int node = blockIdx.x * 8 + (threadIdx.x >> 5);
  int lane = threadIdx.x & 31;
  if (node >= N) return;
  const float* row = g + (size_t)node * F;
  float s = 0.0f, d = 0.0f;
  for (int i = lane; i < F; i += 32) {
    float h = row[i];
    s += h * asrc[i];
    d += h * adst[i];
  }
  #pragma unroll
  for (int off = 16; off > 0; off >>= 1) {
    s += __shfl_down(s, off, 32);
    d += __shfl_down(d, off, 32);
  }
  if (lane == 0) { outS[node] = s; outD[node] = d; }
}

// ---------------------------------------------------------------------------
// Edge helpers. Edge ids [0,E) from edge_index; [E,E+N) are self loops.
// ---------------------------------------------------------------------------
__device__ __forceinline__ void edge_sd(const int* ei, int t, int& s, int& d) {
  if (t < GAT_E) { s = ei[t]; d = ei[GAT_E + t]; }
  else           { s = d = t - GAT_E; }
}
__device__ __forceinline__ unsigned f32_ord(float f) {
  unsigned u = __float_as_uint(f);
  return (u & 0x80000000u) ? ~u : (u | 0x80000000u);
}
__device__ __forceinline__ float f32_deord(unsigned u) {
  u = (u & 0x80000000u) ? (u & 0x7FFFFFFFu) : ~u;
  return __uint_as_float(u);
}

__global__ void gat_deg(const int* __restrict__ ei, int* __restrict__ deg) {
  int t = blockIdx.x * blockDim.x + threadIdx.x;
  if (t >= GAT_ET) return;
  int d = (t < GAT_E) ? ei[GAT_E + t] : (t - GAT_E);
  atomicAdd(&deg[d], 1);
}

// single-block exclusive scan of degrees -> offsets (+ cursor copy)
__global__ void gat_scan(const int* __restrict__ deg, int* __restrict__ offs,
                         int* __restrict__ cursor, int n) {
  __shared__ int sh[1024];
  int carry = 0;
  for (int base = 0; base < n; base += 1024) {
    int i = base + (int)threadIdx.x;
    int v = (i < n) ? deg[i] : 0;
    sh[threadIdx.x] = v;
    __syncthreads();
    for (int off = 1; off < 1024; off <<= 1) {
      int t = (threadIdx.x >= (unsigned)off) ? sh[threadIdx.x - off] : 0;
      __syncthreads();
      sh[threadIdx.x] += t;
      __syncthreads();
    }
    int incl = sh[threadIdx.x];
    if (i < n) { int e = carry + incl - v; offs[i] = e; cursor[i] = e; }
    int total = sh[1023];
    __syncthreads();
    carry += total;
  }
}

__global__ void gat_edge_max(const int* __restrict__ ei,
                             const float* __restrict__ as,
                             const float* __restrict__ ad,
                             float* __restrict__ ebuf,
                             unsigned* __restrict__ maxv) {
  int t = blockIdx.x * blockDim.x + threadIdx.x;
  if (t >= GAT_ET) return;
  int s, d; edge_sd(ei, t, s, d);
  float e = as[s] + ad[d];
  e = (e > 0.0f) ? e : 0.2f * e;               // LeakyReLU(0.2)
  ebuf[t] = e;
  atomicMax(&maxv[d], f32_ord(e));
}

__global__ void gat_edge_exp(const int* __restrict__ ei,
                             float* __restrict__ ebuf,
                             const unsigned* __restrict__ maxv,
                             float* __restrict__ denom) {
  int t = blockIdx.x * blockDim.x + threadIdx.x;
  if (t >= GAT_ET) return;
  int d = (t < GAT_E) ? ei[GAT_E + t] : (t - GAT_E);
  float ex = __expf(ebuf[t] - f32_deord(maxv[d]));
  ebuf[t] = ex;
  atomicAdd(&denom[d], ex);
}

__global__ void gat_fill(const int* __restrict__ ei, int* __restrict__ cursor,
                         const float* __restrict__ ebuf, int* __restrict__ ssrc,
                         float* __restrict__ sex, int* __restrict__ epos) {
  int t = blockIdx.x * blockDim.x + threadIdx.x;
  if (t >= GAT_ET) return;
  int s, d; edge_sd(ei, t, s, d);
  int pos = atomicAdd(&cursor[d], 1);
  ssrc[pos] = s;
  sex[pos]  = ebuf[t];
  epos[t]   = pos;
}

__global__ void gat_scatter(const float* __restrict__ ebuf,
                            const int* __restrict__ epos,
                            float* __restrict__ sex) {
  int t = blockIdx.x * blockDim.x + threadIdx.x;
  if (t >= GAT_ET) return;
  sex[epos[t]] = ebuf[t];
}

// ---------------------------------------------------------------------------
// Aggregation: one block per destination node, one thread per feature.
// Optionally also emits the f16 copy of the activations for the next GEMM.
// ---------------------------------------------------------------------------
__global__ void gat_agg(const int* __restrict__ offs, const int* __restrict__ deg,
                        const float* __restrict__ denom,
                        const int* __restrict__ ssrc, const float* __restrict__ sex,
                        const float* __restrict__ g, const float* __restrict__ bias,
                        float* __restrict__ out, _Float16* __restrict__ out16,
                        int F, int doRelu) {
  int i = blockIdx.x;
  int f = threadIdx.x;
  int beg = offs[i], cnt = deg[i];
  float inv = 1.0f / denom[i];
  float acc = 0.0f;
  for (int e = 0; e < cnt; ++e) {
    int s = ssrc[beg + e];
    float w = sex[beg + e] * inv;
    if (e + 1 < cnt)
      __builtin_prefetch(&g[(size_t)ssrc[beg + e + 1] * F + f], 0, 3);
    acc += w * g[(size_t)s * F + f];
  }
  float v = acc + bias[f];
  if (doRelu) v = fmaxf(v, 0.0f);
  out[(size_t)i * F + f] = v;
  if (out16) out16[(size_t)i * F + f] = (_Float16)v;
}

// ---------------------------------------------------------------------------
extern "C" void kernel_launch(void* const* d_in, const int* in_sizes, int n_in,
                              void* d_out, int out_size, void* d_ws, size_t ws_size,
                              hipStream_t stream) {
  const float* x   = (const float*)d_in[0];
  const int*   ei  = (const int*)  d_in[1];
  const float* W1  = (const float*)d_in[2];
  const float* aS1 = (const float*)d_in[3];
  const float* aD1 = (const float*)d_in[4];
  const float* b1  = (const float*)d_in[5];
  const float* W2  = (const float*)d_in[6];
  const float* aS2 = (const float*)d_in[7];
  const float* aD2 = (const float*)d_in[8];
  const float* b2  = (const float*)d_in[9];

  const int N = GAT_N, Et = GAT_ET;
  float* out_x2 = (float*)d_out;                       // [N, 128] first
  float* out_h  = (float*)d_out + (size_t)N * GAT_OUT; // [N, 256] second

  // carve workspace
  char* w = (char*)d_ws;
  auto carve = [&](size_t bytes) -> void* {
    void* p = (void*)w;
    w += (bytes + 255) & ~(size_t)255;
    return p;
  };
  float*     g1   = (float*)    carve((size_t)N * GAT_HID * 4);
  float*     g2   = (float*)    carve((size_t)N * GAT_OUT * 4);
  _Float16*  Xh   = (_Float16*) carve((size_t)N * GAT_IN * 2);
  _Float16*  Hh   = (_Float16*) carve((size_t)N * GAT_HID * 2);
  _Float16*  W1t  = (_Float16*) carve((size_t)GAT_IN * GAT_HID * 2);
  _Float16*  W2t  = (_Float16*) carve((size_t)GAT_HID * GAT_OUT * 2);
  float*     alS  = (float*)    carve((size_t)N * 4);
  float*     alD  = (float*)    carve((size_t)N * 4);
  float*     ebuf = (float*)    carve((size_t)Et * 4);
  int*       ssrc = (int*)      carve((size_t)Et * 4);
  float*     sex  = (float*)    carve((size_t)Et * 4);
  int*       epos = (int*)      carve((size_t)Et * 4);
  int*       deg  = (int*)      carve((size_t)N * 4);
  int*       offs = (int*)      carve((size_t)N * 4);
  int*       curs = (int*)      carve((size_t)N * 4);
  float*     den  = (float*)    carve((size_t)N * 4);
  unsigned*  maxv = (unsigned*) carve((size_t)N * 4);

  const int eb = (Et + 255) / 256;

  hipMemsetAsync(deg,  0, (size_t)N * 4, stream);
  hipMemsetAsync(den,  0, (size_t)N * 4, stream);
  hipMemsetAsync(maxv, 0, (size_t)N * 4, stream);  // 0 == below all reals in ord-space

  // one-time conversions for this call
  int xn4 = (N * GAT_IN) / 4;
  gat_cvt_f16<<<(xn4 + 255) / 256, 256, 0, stream>>>(x, Xh, xn4);
  gat_cvt_transpose<<<(GAT_IN * GAT_HID + 255) / 256, 256, 0, stream>>>(
      W1, W1t, GAT_IN, GAT_HID);
  gat_cvt_transpose<<<(GAT_HID * GAT_OUT + 255) / 256, 256, 0, stream>>>(
      W2, W2t, GAT_HID, GAT_OUT);

  // ---- layer 1 ----
  dim3 g1grid((N + 127) / 128, GAT_HID / 64);
  gat_gemm_wmma<<<g1grid, 256, 0, stream>>>(Xh, W1t, g1, N, GAT_IN, GAT_HID);
  gat_alpha<<<(N + 7) / 8, 256, 0, stream>>>(g1, aS1, aD1, alS, alD, N, GAT_HID);
  gat_deg<<<eb, 256, 0, stream>>>(ei, deg);
  gat_scan<<<1, 1024, 0, stream>>>(deg, offs, curs, N);
  gat_edge_max<<<eb, 256, 0, stream>>>(ei, alS, alD, ebuf, maxv);
  gat_edge_exp<<<eb, 256, 0, stream>>>(ei, ebuf, maxv, den);
  gat_fill<<<eb, 256, 0, stream>>>(ei, curs, ebuf, ssrc, sex, epos);
  gat_agg<<<N, GAT_HID, 0, stream>>>(offs, deg, den, ssrc, sex, g1, b1,
                                     out_h, Hh, GAT_HID, 1);

  // ---- layer 2 (same CSR structure, new attention weights) ----
  dim3 g2grid((N + 127) / 128, GAT_OUT / 64);
  gat_gemm_wmma<<<g2grid, 256, 0, stream>>>(Hh, W2t, g2, N, GAT_HID, GAT_OUT);
  gat_alpha<<<(N + 7) / 8, 256, 0, stream>>>(g2, aS2, aD2, alS, alD, N, GAT_OUT);
  hipMemsetAsync(den,  0, (size_t)N * 4, stream);
  hipMemsetAsync(maxv, 0, (size_t)N * 4, stream);
  gat_edge_max<<<eb, 256, 0, stream>>>(ei, alS, alD, ebuf, maxv);
  gat_edge_exp<<<eb, 256, 0, stream>>>(ei, ebuf, maxv, den);
  gat_scatter<<<eb, 256, 0, stream>>>(ebuf, epos, sex);
  gat_agg<<<N, GAT_OUT, 0, stream>>>(offs, deg, den, ssrc, sex, g2, b2,
                                     out_x2, (_Float16*)nullptr, GAT_OUT, 0);
}
